// TIPAR_81527069212869
// MI455X (gfx1250) — matrix operand order repaired and checked by reading, not compiled, not run on hardware
//
#include <hip/hip_runtime.h>
#include <hip/hip_bf16.h>

// ---------------------------------------------------------------------------
// Problem constants (fixed by the reference)
// ---------------------------------------------------------------------------
#define N_NODES 100000
#define DIM     256
#define RAW     640
#define M_MSG   200000

typedef __bf16 v16bf __attribute__((ext_vector_type(16)));
typedef __bf16 v8bf  __attribute__((ext_vector_type(8)));
typedef float  v8f   __attribute__((ext_vector_type(8)));
typedef int    v4i   __attribute__((ext_vector_type(4)));

#define WMMA_BF16(A, B, C) \
    __builtin_amdgcn_wmma_f32_16x16x32_bf16(false, (A), false, (B), (short)0, (C), false, false)

// ---------------------------------------------------------------------------
// CDNA5 async global->LDS copy (ASYNCcnt-tracked, no VGPR round trip).
// Builtin signature (from hipcc diagnostic): arg0 = AS(1) v4i* (non-const),
// arg1 = LDS-side pointer, arg2/arg3 = imm offset / cpol.
// Guarded: falls back to synchronous staging if the builtin is absent.
// ---------------------------------------------------------------------------
#if defined(__gfx1250__) && __has_builtin(__builtin_amdgcn_global_load_async_to_lds_b128)
#define HAVE_ASYNC_LDS 1
#else
#define HAVE_ASYNC_LDS 0
#endif

#if HAVE_ASYNC_LDS
__device__ __forceinline__ void async_copy16(const void* g, void* l) {
    __builtin_amdgcn_global_load_async_to_lds_b128(
        (__attribute__((address_space(1))) v4i*)(void*)g,
        (__attribute__((address_space(3))) v4i*)l, 0, 0);
}
__device__ __forceinline__ void wait_async0() {
#if __has_builtin(__builtin_amdgcn_s_wait_asynccnt)
    __builtin_amdgcn_s_wait_asynccnt(0);
#else
    asm volatile("s_wait_asynccnt 0" ::: "memory");
#endif
}
#endif

// non-returning fp32 atomic add -> global_atomic_add_f32 (agg fits in 192MB L2)
__device__ __forceinline__ void atomic_add_f32(float* p, float v) {
    (void)__hip_atomic_fetch_add(p, v, __ATOMIC_RELAXED, __HIP_MEMORY_SCOPE_AGENT);
}

// ---------------------------------------------------------------------------
// Fragment loaders (contiguous k order, same for A and B, wave32)
// A 16x32 bf16: lane L<16 -> row L, K = k0..k0+15
//               lane L>=16 -> row L-16, K = k0+16..k0+31
// ---------------------------------------------------------------------------
__device__ __forceinline__ v16bf lds_a_frag(const __bf16* base, int stride, int k0, int lane) {
    int m  = lane & 15;
    int kb = k0 + ((lane >> 4) << 4);
    const __bf16* p = base + m * stride + kb;
    v8bf lo = *(const v8bf*)p;
    v8bf hi = *(const v8bf*)(p + 8);
    v16bf a;
#pragma unroll
    for (int i = 0; i < 8; ++i) { a[i] = lo[i]; a[i + 8] = hi[i]; }
    return a;
}

// B 32x16 bf16 from a column-major (transposed) weight WT[c*K + k]:
// lane L<16 -> col L, K = k0..k0+15 ; lane L>=16 -> col L-16, K = k0+16..k0+31
__device__ __forceinline__ v16bf g_b_frag(const __bf16* WT, int K, int colbase, int k0, int lane) {
    int c  = colbase + (lane & 15);
    int kk = k0 + ((lane >> 4) << 4);
    return *(const v16bf*)(WT + (long)c * K + kk);
}

// ---------------------------------------------------------------------------
// Kernel: zero a float region (counts + agg accumulators)
// ---------------------------------------------------------------------------
__global__ void k_zero_f32(float* p, long n) {
    long stride = (long)gridDim.x * blockDim.x;
    for (long i = (long)blockIdx.x * blockDim.x + threadIdx.x; i < n; i += stride)
        p[i] = 0.0f;
}

// ---------------------------------------------------------------------------
// Kernel: convert fp32 weight W[K,C] (row-major) to bf16 transposed WT[c*K+k]
// ---------------------------------------------------------------------------
__global__ void k_cvtT_bf16(const float* __restrict__ W, __bf16* __restrict__ WT, int K, int C) {
    long n = (long)K * C;
    long stride = (long)gridDim.x * blockDim.x;
    for (long i = (long)blockIdx.x * blockDim.x + threadIdx.x; i < n; i += stride) {
        long k = i / C, c = i % C;
        WT[c * (long)K + k] = (__bf16)W[i];
    }
}

// ---------------------------------------------------------------------------
// Kernel: message projection GEMM (bf16 WMMA) + atomic scatter-mean pieces.
// One block = 128 threads (4 waves) handles a 16-row tile of one store.
// The 16 message rows are CONSECUTIVE -> the 16x640 fp32 tile is one
// contiguous 40KB global block: ideal for global_load_async_to_lds_b128.
// ---------------------------------------------------------------------------
__global__ __launch_bounds__(128) void k_msg_gemm_scatter(
    const float* __restrict__ s_raw, const float* __restrict__ d_raw,
    const int* __restrict__ s_src,   const int* __restrict__ d_src,
    const float* __restrict__ b_src, const float* __restrict__ b_dst,
    const __bf16* __restrict__ WsrcT, const __bf16* __restrict__ WdstT,
    float* __restrict__ agg, float* __restrict__ counts)
{
    __shared__ __align__(32) __bf16 Abf[16 * RAW];   // 20 KB: 16 raw rows as bf16
#if HAVE_ASYNC_LDS
    __shared__ __align__(16) float  Af32[16 * RAW];  // 40 KB: async landing pad
#endif

    const int tile    = blockIdx.x;
    const int tiles_s = M_MSG / 16;
    const bool is_d   = tile >= tiles_s;
    const float* raw  = is_d ? d_raw : s_raw;
    const int* srcp   = is_d ? d_src : s_src;
    const float* bias = is_d ? b_dst : b_src;
    const __bf16* WT  = is_d ? WdstT : WsrcT;
    const int rowbase = (is_d ? tile - tiles_s : tile) * 16;

    // per-node message counters: exactly one increment per row
    if (threadIdx.x < 16)
        atomic_add_f32(&counts[srcp[rowbase + threadIdx.x]], 1.0f);

#if HAVE_ASYNC_LDS
    {   // async DMA the contiguous 40KB fp32 tile into LDS
        const char* gsrc = (const char*)(raw + (long)rowbase * RAW);
        char*       lds  = (char*)Af32;
        for (int i = threadIdx.x; i < (16 * RAW) / 4; i += blockDim.x)
            async_copy16(gsrc + (size_t)i * 16, lds + (size_t)i * 16);
        wait_async0();
        __syncthreads();
        for (int i = threadIdx.x; i < 16 * RAW; i += blockDim.x)
            Abf[i] = (__bf16)Af32[i];
    }
#else
    for (int i = threadIdx.x; i < 16 * RAW; i += blockDim.x) {
        int m = i / RAW, k = i - m * RAW;
        Abf[i] = (__bf16)raw[(long)(rowbase + m) * RAW + k];
    }
#endif
    __syncthreads();

    const int wave = threadIdx.x >> 5;
    const int lane = threadIdx.x & 31;

    v8f acc[4] = {};
#pragma unroll 4
    for (int k0 = 0; k0 < RAW; k0 += 32) {
        v16bf a = lds_a_frag(Abf, RAW, k0, lane);
#pragma unroll
        for (int t = 0; t < 4; ++t) {
            v16bf b = g_b_frag(WT, RAW, wave * 64 + t * 16, k0, lane);
            acc[t] = WMMA_BF16(a, b, acc[t]);
        }
    }

    // rows owned by this lane in the C/D layout: m = r (+8 for upper lanes)
    int nodes[8];
#pragma unroll
    for (int r = 0; r < 8; ++r)
        nodes[r] = srcp[rowbase + r + ((lane >> 4) << 3)];

#pragma unroll
    for (int t = 0; t < 4; ++t) {
        int col  = wave * 64 + t * 16 + (lane & 15);
        float bv = bias[col];
#pragma unroll
        for (int r = 0; r < 8; ++r)
            atomic_add_f32(&agg[(long)nodes[r] * DIM + col], acc[t][r] + bv);
    }
}

// ---------------------------------------------------------------------------
// Kernel: GRU update for 16 nodes per block (256 threads, 8 waves).
// r/z gates: one accumulator gets BOTH agg@W and mem@U. n gate: xn/hn kept
// separate (r * hn coupling). 12 WMMAs per K-step per wave, 96 WMMAs/wave.
// agg/memory tiles are contiguous 16KB blocks -> async DMA into LDS.
// ---------------------------------------------------------------------------
__global__ __launch_bounds__(256) void k_gru(
    const float* __restrict__ aggsum, const float* __restrict__ counts,
    const float* __restrict__ memory,
    const __bf16* __restrict__ WgT, const __bf16* __restrict__ UgT,
    const float* __restrict__ b_gru, float* __restrict__ out)
{
    __shared__ __align__(32) __bf16 Abf[16 * DIM];  // agg mean, bf16 (8 KB)
    __shared__ __align__(32) __bf16 Mbf[16 * DIM];  // memory,   bf16 (8 KB)
    __shared__ __align__(32) float  Mf [16 * DIM];  // memory,   fp32 (16 KB)
#if HAVE_ASYNC_LDS
    __shared__ __align__(16) float  Ag [16 * DIM];  // agg sum,  fp32 (16 KB)
#endif
    __shared__ float cnt[16];

    const int n0 = blockIdx.x * 16;

    if (threadIdx.x < 16) cnt[threadIdx.x] = counts[n0 + threadIdx.x];

#if HAVE_ASYNC_LDS
    {   // async DMA the two contiguous 16KB fp32 tiles into LDS
        const char* ga = (const char*)(aggsum + (long)n0 * DIM);
        const char* gm = (const char*)(memory + (long)n0 * DIM);
        for (int i = threadIdx.x; i < (16 * DIM) / 4; i += blockDim.x) {
            async_copy16(ga + (size_t)i * 16, (char*)Ag + (size_t)i * 16);
            async_copy16(gm + (size_t)i * 16, (char*)Mf + (size_t)i * 16);
        }
        wait_async0();
    }
    __syncthreads();
    for (int i = threadIdx.x; i < 16 * DIM; i += blockDim.x) {
        int m = i >> 8;
        float inv = 1.0f / fmaxf(cnt[m], 1.0f);
        Abf[i] = (__bf16)(Ag[i] * inv);
        Mbf[i] = (__bf16)Mf[i];
    }
#else
    __syncthreads();
    for (int i = threadIdx.x; i < 16 * DIM; i += blockDim.x) {
        int m = i >> 8, c = i & 255;
        float inv = 1.0f / fmaxf(cnt[m], 1.0f);
        float av  = aggsum[(long)(n0 + m) * DIM + c] * inv;
        float mv  = memory[(long)(n0 + m) * DIM + c];
        Abf[i] = (__bf16)av;
        Mbf[i] = (__bf16)mv;
        Mf[i]  = mv;
    }
#endif
    __syncthreads();

    const int wave = threadIdx.x >> 5;
    const int lane = threadIdx.x & 31;

    v8f Sr[2] = {}, Sz[2] = {}, Xn[2] = {}, Hn[2] = {};

#pragma unroll 2
    for (int k0 = 0; k0 < DIM; k0 += 32) {
        v16bf aA = lds_a_frag(Abf, DIM, k0, lane);
        v16bf aM = lds_a_frag(Mbf, DIM, k0, lane);
#pragma unroll
        for (int t = 0; t < 2; ++t) {
            int c = wave * 32 + t * 16;
            v16bf bWr = g_b_frag(WgT, DIM, c,       k0, lane);
            v16bf bUr = g_b_frag(UgT, DIM, c,       k0, lane);
            v16bf bWz = g_b_frag(WgT, DIM, 256 + c, k0, lane);
            v16bf bUz = g_b_frag(UgT, DIM, 256 + c, k0, lane);
            v16bf bWn = g_b_frag(WgT, DIM, 512 + c, k0, lane);
            v16bf bUn = g_b_frag(UgT, DIM, 512 + c, k0, lane);
            Sr[t] = WMMA_BF16(aA, bWr, Sr[t]);
            Sr[t] = WMMA_BF16(aM, bUr, Sr[t]);
            Sz[t] = WMMA_BF16(aA, bWz, Sz[t]);
            Sz[t] = WMMA_BF16(aM, bUz, Sz[t]);
            Xn[t] = WMMA_BF16(aA, bWn, Xn[t]);
            Hn[t] = WMMA_BF16(aM, bUn, Hn[t]);
        }
    }

#pragma unroll
    for (int t = 0; t < 2; ++t) {
        int col  = wave * 32 + t * 16 + (lane & 15);
        float br = b_gru[col];
        float bz = b_gru[256 + col];
        float bn = b_gru[512 + col];
#pragma unroll
        for (int r = 0; r < 8; ++r) {
            int   m   = r + ((lane >> 4) << 3);
            float rr  = 1.0f / (1.0f + __expf(-(Sr[t][r] + br)));
            float zz  = 1.0f / (1.0f + __expf(-(Sz[t][r] + bz)));
            float nn  = tanhf(Xn[t][r] + bn + rr * Hn[t][r]);
            float mem = Mf[m * DIM + col];
            float h   = (1.0f - zz) * nn + zz * mem;
            out[(long)(n0 + m) * DIM + col] = (cnt[m] > 0.0f) ? h : mem;
        }
    }
}

// ---------------------------------------------------------------------------
// Host launcher
// ---------------------------------------------------------------------------
extern "C" void kernel_launch(void* const* d_in, const int* in_sizes, int n_in,
                              void* d_out, int out_size, void* d_ws, size_t ws_size,
                              hipStream_t stream) {
    const float* s_raw  = (const float*)d_in[0];
    const float* d_raw  = (const float*)d_in[1];
    const int*   s_src  = (const int*)d_in[2];
    const int*   d_src  = (const int*)d_in[3];
    // d_in[4], d_in[5] (timestamps) only feed the unused t_max -> skipped
    const float* W_src  = (const float*)d_in[6];
    const float* b_src  = (const float*)d_in[7];
    const float* W_dst  = (const float*)d_in[8];
    const float* b_dst  = (const float*)d_in[9];
    const float* W_gru  = (const float*)d_in[10];
    const float* U_gru  = (const float*)d_in[11];
    const float* b_gru  = (const float*)d_in[12];
    const float* memory = (const float*)d_in[13];
    float* out = (float*)d_out;

    // workspace carve-up (256B aligned)
    size_t off = 0;
    auto carve = [&](size_t bytes) { size_t o = off; off += (bytes + 255) & ~(size_t)255; return o; };
    char* ws = (char*)d_ws;
    __bf16* WsrcT = (__bf16*)(ws + carve((size_t)RAW * DIM * 2));      // [256][640]
    __bf16* WdstT = (__bf16*)(ws + carve((size_t)RAW * DIM * 2));
    __bf16* WgT   = (__bf16*)(ws + carve((size_t)DIM * 3 * DIM * 2));  // [768][256]
    __bf16* UgT   = (__bf16*)(ws + carve((size_t)DIM * 3 * DIM * 2));
    float*  counts = (float*)(ws + carve((size_t)N_NODES * 4));
    float*  agg    = (float*)(ws + carve((size_t)N_NODES * DIM * 4));  // 102.4 MB (fits L2)

    // 1) zero counts + agg (contiguous region)
    {
        long nzero = ((char*)(agg + (long)N_NODES * DIM) - (char*)counts) / 4;
        k_zero_f32<<<2048, 256, 0, stream>>>(counts, nzero);
    }
    // 2) convert + transpose weights to bf16
    k_cvtT_bf16<<<512, 256, 0, stream>>>(W_src, WsrcT, RAW, DIM);
    k_cvtT_bf16<<<512, 256, 0, stream>>>(W_dst, WdstT, RAW, DIM);
    k_cvtT_bf16<<<512, 256, 0, stream>>>(W_gru, WgT,   DIM, 3 * DIM);
    k_cvtT_bf16<<<512, 256, 0, stream>>>(U_gru, UgT,   DIM, 3 * DIM);
    // 3) message GEMM + scatter (both stores, 2M/16 row tiles)
    k_msg_gemm_scatter<<<2 * (M_MSG / 16), 128, 0, stream>>>(
        s_raw, d_raw, s_src, d_src, b_src, b_dst, WsrcT, WdstT, agg, counts);
    // 4) GRU memory update, masked writeback
    k_gru<<<N_NODES / 16, 256, 0, stream>>>(agg, counts, memory, WgT, UgT, b_gru, out);
}
